// LSTMModel_4105988735599
// MI455X (gfx1250) — matrix operand order repaired and checked by reading, not compiled, not run on hardware
//
#include <hip/hip_runtime.h>
#include <math.h>

// ---------------------------------------------------------------------------
// 2-layer LSTM + linear head for MI455X (gfx1250, wave32, WMMA bf16 + TDM).
//   B=32, T=1024, IN=256, H=768 (4H=3072), OUT=128, M=B*T=32768
//
//   * input-side GEMMs batched over all T: WMMA bf16, 1 M-tile x 4 N-tiles
//     per wave (A-fragment reuse, fewer vmem per wmma)
//   * recurrence: 1 launch per timestep, 48 blocks x 8 waves; h_prev staged
//     into LDS via TENSOR_LOAD_TO_LDS (TDM, s_wait_tensorcnt), gates staged
//     in LDS, fused sigmoid/tanh cell update. W_hh (4.7MB bf16) L2-resident.
// ---------------------------------------------------------------------------

typedef __bf16 bf16;
typedef unsigned int u32;
typedef unsigned long long u64;
typedef __attribute__((ext_vector_type(16))) __bf16 v16bf;
typedef __attribute__((ext_vector_type(8)))  __bf16 v8bf;
typedef __attribute__((ext_vector_type(8)))  float  v8f;
typedef __attribute__((ext_vector_type(4)))  unsigned int v4u;
typedef __attribute__((ext_vector_type(8)))  int v8i;
typedef __attribute__((ext_vector_type(4)))  int v4i;

#define B_   32
#define T_   1024
#define IN_  256
#define H_   768
#define G4_  3072
#define OUT_ 128
#define M_   32768   // B_*T_

#if defined(__has_builtin)
#if __has_builtin(__builtin_amdgcn_tensor_load_to_lds) && \
    __has_builtin(__builtin_amdgcn_s_wait_tensorcnt)
#define USE_TDM 1
#endif
#endif

// ---- WMMA fragment loaders (ISA 7.12.2 layouts, wave32) -------------------
// A (16x32 bf16, MxK): lanes 0-15 -> M=lane, K = {0..7, 16..23}
//                      lanes 16-31 -> M=lane-16, K = {8..15, 24..31}
__device__ __forceinline__ v16bf load_a_frag(const bf16* __restrict__ A,
                                             int lda, int mBase, int kBase) {
  int lane = threadIdx.x & 31;
  int m  = mBase + (lane & 15);
  int kb = kBase + ((lane >> 4) << 3);          // +0 or +8
  const v8bf* p0 = (const v8bf*)(A + (size_t)m * lda + kb);
  const v8bf* p1 = (const v8bf*)(A + (size_t)m * lda + kb + 16);
  v8bf lo = *p0, hi = *p1;
  v16bf r;
#pragma unroll
  for (int i = 0; i < 8; ++i) { r[i] = lo[i]; r[i + 8] = hi[i]; }
  return r;
}

// B (32x16 bf16, KxN) with B[k][n] = W[n][k], W row-major [N,K]:
// lanes 0-15 -> N=lane, K={0..7,16..23} ; lanes 16-31 -> N=lane-16, K={8..15,24..31}
__device__ __forceinline__ v16bf load_b_frag(const bf16* __restrict__ W,
                                             int ldw, int nBase, int kBase) {
  int lane = threadIdx.x & 31;
  int n = nBase + (lane & 15);
  int k = kBase + ((lane >> 4) << 3);           // +0 or +8
  const v8bf* p = (const v8bf*)(W + (size_t)n * ldw + k);
  v8bf lo = p[0], hi = p[2];
  v16bf r;
#pragma unroll
  for (int i = 0; i < 8; ++i) { r[i] = lo[i]; r[i + 8] = hi[i]; }
  return r;
}

__device__ __forceinline__ v8f wmma_bf16(v16bf a, v16bf b, v8f c) {
  return __builtin_amdgcn_wmma_f32_16x16x32_bf16(false, a, false, b,
                                                 (short)0, c, false, false);
}

// ---- generic WMMA GEMM: C[M,N] = A[M,K] @ W[N,K]^T + bias[N] --------------
// each wave: 1 M-tile x NT N-tiles (A-fragment reused NT times)
template <bool OUT_BF16, int NT>
__global__ __launch_bounds__(256)
void gemm_wmma(const bf16* __restrict__ A, const bf16* __restrict__ W,
               const float* __restrict__ bias, void* __restrict__ Cout,
               int M, int N, int K) {
  int wave    = blockIdx.x * 8 + (threadIdx.x >> 5);
  int nGroups = N / (16 * NT);
  int m0 = (wave / nGroups) << 4;
  int ng = (wave % nGroups) * (16 * NT);
  int lane = threadIdx.x & 31;

  v8f acc[NT];
#pragma unroll
  for (int i = 0; i < NT; ++i) acc[i] = (v8f){};

  for (int k = 0; k < K; k += 32) {
    if (k + 32 < K)  // pull next W slab (global_prefetch_b8)
      __builtin_prefetch(W + (size_t)(ng + (lane & 15)) * K + k + 32, 0, 0);
    v16bf a = load_a_frag(A, K, m0, k);
#pragma unroll
    for (int i = 0; i < NT; ++i) {
      v16bf b = load_b_frag(W, K, ng + 16 * i, k);
      acc[i] = wmma_bf16(a, b, acc[i]);
    }
  }

  // C/D layout: vgpr v, lanes 0-15 -> M=v, N=lane ; lanes 16-31 -> M=v+8
  int rowB = m0 + ((lane >> 4) << 3);
#pragma unroll
  for (int i = 0; i < NT; ++i) {
    int col  = ng + 16 * i + (lane & 15);
    float bv = bias ? bias[col] : 0.0f;
#pragma unroll
    for (int v = 0; v < 8; ++v) {
      size_t idx = (size_t)(rowB + v) * N + col;
      float r = acc[i][v] + bv;
      if (OUT_BF16) ((bf16*)Cout)[idx] = (bf16)r;
      else          ((float*)Cout)[idx] = r;
    }
  }
}

// ---- fused LSTM recurrent step --------------------------------------------
// grid = H/16 = 48 blocks, 256 threads = 8 waves; wave = (gate<<1)|mTile.
// h_prev (32x768 bf16 = 48KB) staged into LDS once per block via TDM.
__global__ __launch_bounds__(256)
void lstm_step(const bf16* __restrict__ xg,    // [M,4H], row = b*T + t
               const bf16* __restrict__ Whh,   // [4H,H] row-major
               const bf16* __restrict__ h_in,  // [B,H]
               bf16* __restrict__ h_out,       // [B,H]
               float* __restrict__ c_state,    // [B,H]
               bf16* __restrict__ seq_out,     // [B,T,H]
               int t) {
  __shared__ bf16  hsh[B_][H_];       // 48 KB
  __shared__ float gbuf[4][B_][16];   //  8 KB

#ifdef USE_TDM
  if (threadIdx.x < 32) {             // wave 0 issues one TDM op per block
    u32 lds = (u32)(size_t)&hsh[0][0];
    u64 ga  = (u64)(size_t)h_in;
    // D# group0: count=1 | lds_addr | global_addr | type=2 (ISA 8.3)
    v4u g0;
    g0[0] = 1u;                                   // count=1, gather off
    g0[1] = lds;                                  // lds_addr [63:32]
    g0[2] = (u32)ga;                              // global_addr [95:64]
    g0[3] = ((u32)(ga >> 32) & 0x01FFFFFFu) | (2u << 30);  // [120:96], type=2
    // D# group1: data_size=2B, tensor 768x32, tile 768x32, stride0=768 (8.4)
    v8i g1;
    g1[0] = (int)(1u << 16);                      // data_size=1 (2 bytes)
    g1[1] = (int)((u32)H_ << 16);                 // tensor_dim0[15:0] @ [63:48]
    g1[2] = (int)((u32)B_ << 16);                 // tensor_dim1[15:0] @ [111:80]
    g1[3] = (int)((u32)H_ << 16);                 // tile_dim0 @ [127:112]
    g1[4] = (int)B_;                              // tile_dim1 @ [143:128]
    g1[5] = (int)H_;                              // tensor_dim0_stride [207:160]
    g1[6] = 0;
    g1[7] = 0;
    v4i gz4 = (v4i){};                            // dims 2-4 unused
    v8i gz8 = (v8i){};
    __builtin_amdgcn_tensor_load_to_lds(g0, g1, gz4, gz4, gz8, 0);
    __builtin_amdgcn_s_wait_tensorcnt(0);
  }
  __syncthreads();
#else
  {  // cooperative fallback: 3072 x 16B chunks over 256 threads
    const ulonglong2* src = (const ulonglong2*)h_in;
    ulonglong2* dst = (ulonglong2*)&hsh[0][0];
    for (int i = threadIdx.x; i < (B_ * H_) / 8; i += 256) dst[i] = src[i];
  }
  __syncthreads();
#endif

  int wave  = threadIdx.x >> 5;
  int lane  = threadIdx.x & 31;
  int mTile = wave & 1;            // batches 0-15 / 16-31
  int gate  = wave >> 1;           // i,f,g,o
  int jBase = blockIdx.x << 4;
  int n0    = gate * H_ + jBase;

  v8f acc = {};
#pragma unroll 4
  for (int k = 0; k < H_; k += 32) {
    if (k + 32 < H_)
      __builtin_prefetch(Whh + (size_t)(n0 + (lane & 15)) * H_ + k + 32, 0, 0);
    // A fragment from LDS (ds_load_b128)
    v16bf a;
    {
      int m  = (mTile << 4) + (lane & 15);
      int kb = k + ((lane >> 4) << 3);
      v8bf lo = *(const v8bf*)&hsh[m][kb];
      v8bf hi = *(const v8bf*)&hsh[m][kb + 16];
#pragma unroll
      for (int i = 0; i < 8; ++i) { a[i] = lo[i]; a[i + 8] = hi[i]; }
    }
    v16bf b = load_b_frag(Whh, H_, n0, k);
    acc = wmma_bf16(a, b, acc);
  }

  // add precomputed input-gate activations, stash per-gate tile in LDS
  int j     = lane & 15;
  int mBase = (mTile << 4) + ((lane >> 4) << 3);
#pragma unroll
  for (int v = 0; v < 8; ++v) {
    int bb  = mBase + v;
    float xv = (float)xg[(size_t)(bb * T_ + t) * G4_ + n0 + j];
    gbuf[gate][bb][j] = acc[v] + xv;
  }
  __syncthreads();

  // cell update: 512 (b,j) pairs over 256 threads
  for (int idx = threadIdx.x; idx < B_ * 16; idx += 256) {
    int bb = idx >> 4, jj = idx & 15;
    int jg = jBase + jj;
    float ig = gbuf[0][bb][jj];
    float fg = gbuf[1][bb][jj];
    float gg = gbuf[2][bb][jj];
    float og = gbuf[3][bb][jj];
    ig = 1.0f / (1.0f + __expf(-ig));
    fg = 1.0f / (1.0f + __expf(-fg));
    og = 1.0f / (1.0f + __expf(-og));
    gg = tanhf(gg);
    float c = fg * c_state[bb * H_ + jg] + ig * gg;
    float h = og * tanhf(c);
    c_state[bb * H_ + jg] = c;
    h_out[bb * H_ + jg] = (bf16)h;
    seq_out[((size_t)bb * T_ + t) * H_ + jg] = (bf16)h;
  }
}

// ---- small utility kernels ------------------------------------------------
__global__ void cvt_f32_bf16(const float* __restrict__ in, bf16* __restrict__ out, int n) {
  int i = blockIdx.x * blockDim.x + threadIdx.x;
  if (i < n) out[i] = (bf16)in[i];
}
__global__ void sum_bias(const float* __restrict__ a, const float* __restrict__ b,
                         float* __restrict__ out, int n) {
  int i = blockIdx.x * blockDim.x + threadIdx.x;
  if (i < n) out[i] = a[i] + b[i];
}
__global__ void zero_bf16(bf16* p, int n) {
  int i = blockIdx.x * blockDim.x + threadIdx.x;
  if (i < n) p[i] = (bf16)0.0f;
}
__global__ void zero_f32(float* p, int n) {
  int i = blockIdx.x * blockDim.x + threadIdx.x;
  if (i < n) p[i] = 0.0f;
}

// ---------------------------------------------------------------------------
extern "C" void kernel_launch(void* const* d_in, const int* in_sizes, int n_in,
                              void* d_out, int out_size, void* d_ws, size_t ws_size,
                              hipStream_t stream) {
  const float* x    = (const float*)d_in[0];
  const float* Wih0 = (const float*)d_in[1];
  const float* Whh0 = (const float*)d_in[2];
  const float* bih0 = (const float*)d_in[3];
  const float* bhh0 = (const float*)d_in[4];
  const float* Wih1 = (const float*)d_in[5];
  const float* Whh1 = (const float*)d_in[6];
  const float* bih1 = (const float*)d_in[7];
  const float* bhh1 = (const float*)d_in[8];
  const float* Wlin = (const float*)d_in[9];
  const float* blin = (const float*)d_in[10];
  float* out = (float*)d_out;

  // workspace carve-up (256B aligned)
  char* ws = (char*)d_ws;
  auto carve = [&](size_t bytes) {
    char* p = ws; ws += (bytes + 255) & ~(size_t)255; return p;
  };
  bf16*  xb     = (bf16*)carve((size_t)M_ * IN_ * 2);
  bf16*  wih0b  = (bf16*)carve((size_t)G4_ * IN_ * 2);
  bf16*  whh0b  = (bf16*)carve((size_t)G4_ * H_ * 2);
  bf16*  wih1b  = (bf16*)carve((size_t)G4_ * H_ * 2);
  bf16*  whh1b  = (bf16*)carve((size_t)G4_ * H_ * 2);
  bf16*  wlinb  = (bf16*)carve((size_t)OUT_ * H_ * 2);
  float* bias0  = (float*)carve((size_t)G4_ * 4);
  float* bias1  = (float*)carve((size_t)G4_ * 4);
  bf16*  xg     = (bf16*)carve((size_t)M_ * G4_ * 2);   // reused by both layers
  bf16*  h0seq  = (bf16*)carve((size_t)M_ * H_ * 2);
  bf16*  h1seq  = (bf16*)carve((size_t)M_ * H_ * 2);
  bf16*  hping  = (bf16*)carve((size_t)B_ * H_ * 2);
  bf16*  hpong  = (bf16*)carve((size_t)B_ * H_ * 2);
  float* cstate = (float*)carve((size_t)B_ * H_ * 4);

  auto cvt = [&](const float* src, bf16* dst, int n) {
    cvt_f32_bf16<<<(n + 255) / 256, 256, 0, stream>>>(src, dst, n);
  };
  cvt(x,    xb,    M_ * IN_);
  cvt(Wih0, wih0b, G4_ * IN_);
  cvt(Whh0, whh0b, G4_ * H_);
  cvt(Wih1, wih1b, G4_ * H_);
  cvt(Whh1, whh1b, G4_ * H_);
  cvt(Wlin, wlinb, OUT_ * H_);
  sum_bias<<<(G4_ + 255) / 256, 256, 0, stream>>>(bih0, bhh0, bias0, G4_);
  sum_bias<<<(G4_ + 255) / 256, 256, 0, stream>>>(bih1, bhh1, bias1, G4_);

  // --- layer 0: input GEMM xg = x @ Wih0^T + bias0 --------------------------
  {
    int blocks = (M_ / 16) * (G4_ / 64) / 8;   // 12288 (NT=4)
    gemm_wmma<true, 4><<<blocks, 256, 0, stream>>>(xb, wih0b, bias0, xg, M_, G4_, IN_);
  }
  zero_bf16<<<(B_ * H_ + 255) / 256, 256, 0, stream>>>(hping, B_ * H_);
  zero_f32 <<<(B_ * H_ + 255) / 256, 256, 0, stream>>>(cstate, B_ * H_);
  for (int t = 0; t < T_; ++t) {
    const bf16* hin = (t & 1) ? hpong : hping;
    bf16*       hot = (t & 1) ? hping : hpong;
    lstm_step<<<H_ / 16, 256, 0, stream>>>(xg, whh0b, hin, hot, cstate, h0seq, t);
  }

  // --- layer 1: input GEMM xg = h0seq @ Wih1^T + bias1 ----------------------
  {
    int blocks = (M_ / 16) * (G4_ / 64) / 8;
    gemm_wmma<true, 4><<<blocks, 256, 0, stream>>>(h0seq, wih1b, bias1, xg, M_, G4_, H_);
  }
  zero_bf16<<<(B_ * H_ + 255) / 256, 256, 0, stream>>>(hping, B_ * H_);
  zero_f32 <<<(B_ * H_ + 255) / 256, 256, 0, stream>>>(cstate, B_ * H_);
  for (int t = 0; t < T_; ++t) {
    const bf16* hin = (t & 1) ? hpong : hping;
    bf16*       hot = (t & 1) ? hping : hpong;
    lstm_step<<<H_ / 16, 256, 0, stream>>>(xg, whh1b, hin, hot, cstate, h1seq, t);
  }

  // --- head: out = h1seq @ Wlin^T + blin (fp32 out) -------------------------
  {
    int blocks = (M_ / 16) * (OUT_ / 64) / 8;  // 512 (NT=4)
    gemm_wmma<false, 4><<<blocks, 256, 0, stream>>>(h1seq, wlinb, blin, out, M_, OUT_, H_);
  }
}
